// GromovWassersteinLoss_74560632259137
// MI455X (gfx1250) — matrix operand order repaired and checked by reading, not compiled, not run on hardware
//
#include <hip/hip_runtime.h>

#define BDIM 2048
#define NITER 50
#define NNORM 5

typedef __bf16 bf16_t;
typedef __attribute__((ext_vector_type(16))) __bf16 v16bf;
typedef __attribute__((ext_vector_type(8)))  __bf16 v8bf;
typedef __attribute__((ext_vector_type(8)))  float  v8f;

static __device__ __forceinline__ bf16_t f2bf(float f) {
    union { float f; unsigned u; } a; a.f = f;
    unsigned r = a.u + 0x7FFFu + ((a.u >> 16) & 1u);   // round-to-nearest-even
    unsigned short h = (unsigned short)(r >> 16);
    union { unsigned short s; bf16_t b; } o; o.s = h;
    return o.b;
}

// CDNA5 async global->LDS copy (tracked by ASYNCcnt), no VGPR round-trip.
static __device__ __forceinline__ void async_load_b128(unsigned lds_off, const void* gaddr) {
    asm volatile("global_load_async_to_lds_b128 %0, %1, off"
                 :: "v"(lds_off), "v"(gaddr) : "memory");
}
static __device__ __forceinline__ void wait_asynccnt0() {
    asm volatile("s_wait_asynccnt 0" ::: "memory");
}

// ---------------------------------------------------------------------------
// Tiled bf16 GEMM, f32 accum:  C[M,N] = A[M,K] * Bt[N,K]^T   (both row-major).
// 256 threads = 8 wave32s. Block tile 128x128, K-step 32, double-buffered LDS
// staged via global_load_async_to_lds_b128. Each wave: 2x4 16x16 WMMA accs.
// ---------------------------------------------------------------------------
#define LDSP 48   // padded LDS row stride (elements); 96B keeps 16B alignment

static __device__ __forceinline__ void stage_tiles(
        bf16_t* __restrict__ As, bf16_t* __restrict__ Bs,
        const bf16_t* __restrict__ A, const bf16_t* __restrict__ Bt,
        int bm, int bn, int k0, int lda, int ldbt, int tid) {
#pragma unroll
    for (int h = 0; h < 2; ++h) {
        const int c = tid + h * 256;          // 512 16B chunks per 128x32 tile
        const int row  = c >> 2;
        const int col8 = (c & 3) << 3;
        async_load_b128((unsigned)(size_t)(As + row * LDSP + col8),
                        A  + (size_t)(bm + row) * lda  + k0 + col8);
        async_load_b128((unsigned)(size_t)(Bs + row * LDSP + col8),
                        Bt + (size_t)(bn + row) * ldbt + k0 + col8);
    }
}

__global__ __launch_bounds__(256)
void wmma_gemm_bf16(const bf16_t* __restrict__ A, const bf16_t* __restrict__ Bt,
                    float* __restrict__ C, int K, int lda, int ldbt, int ldc) {
    __shared__ bf16_t As[2][128 * LDSP];   // [row][k]
    __shared__ bf16_t Bs[2][128 * LDSP];   // [n][k]
    const int tid   = threadIdx.x;
    const int lane  = tid & 31;
    const int wid   = tid >> 5;
    const int waveM = wid >> 1;            // 0..3 -> 32-row strip
    const int waveN = wid & 1;             // 0..1 -> 64-col strip
    const int bm = blockIdx.y * 128;
    const int bn = blockIdx.x * 128;
    const int lr = lane & 15;

    v8f acc[2][4];
#pragma unroll
    for (int i = 0; i < 2; ++i)
#pragma unroll
        for (int j = 0; j < 4; ++j)
            acc[i][j] = (v8f){0.f, 0.f, 0.f, 0.f, 0.f, 0.f, 0.f, 0.f};

    stage_tiles(As[0], Bs[0], A, Bt, bm, bn, 0, lda, ldbt, tid);
    wait_asynccnt0();
    __syncthreads();

    int cur = 0;
    for (int k0 = 0; k0 < K; k0 += 32) {
        const int nxt = cur ^ 1;
        if (k0 + 32 < K)   // prefetch next K-step into the other buffer (async)
            stage_tiles(As[nxt], Bs[nxt], A, Bt, bm, bn, k0 + 32, lda, ldbt, tid);

        // A fragment: lane<16 -> row M=lr, K in {0..7,16..23}; lane>=16 -> K in {8..15,24..31}
        const int kba = (lane < 16) ? 0 : 8;
        v16bf afrag[2], bfrag[4];
#pragma unroll
        for (int sm = 0; sm < 2; ++sm) {
            const bf16_t* p = As[cur] + (waveM * 32 + sm * 16 + lr) * LDSP + kba;
            v8bf lo = *(const v8bf*)(p);
            v8bf hi = *(const v8bf*)(p + 16);
            afrag[sm] = __builtin_shufflevector(lo, hi, 0,1,2,3,4,5,6,7,8,9,10,11,12,13,14,15);
        }
        // B fragment: lane = column = row of Bt; lane<16 -> K in {0..7,16..23}; lane>=16 -> K in {8..15,24..31}
        const int kbb = (lane < 16) ? 0 : 8;
#pragma unroll
        for (int sn = 0; sn < 4; ++sn) {
            const bf16_t* p = Bs[cur] + (waveN * 64 + sn * 16 + lr) * LDSP + kbb;
            v8bf lo = *(const v8bf*)(p);
            v8bf hi = *(const v8bf*)(p + 16);
            bfrag[sn] = __builtin_shufflevector(lo, hi, 0,1,2,3,4,5,6,7,8,9,10,11,12,13,14,15);
        }
#pragma unroll
        for (int sm = 0; sm < 2; ++sm)
#pragma unroll
            for (int sn = 0; sn < 4; ++sn)
                acc[sm][sn] = __builtin_amdgcn_wmma_f32_16x16x32_bf16(
                    false, afrag[sm], false, bfrag[sn], (short)0, acc[sm][sn],
                    false, false);

        wait_asynccnt0();   // next tiles landed
        __syncthreads();    // everyone done reading cur, data in nxt visible
        cur = nxt;
    }

    // Store: C/D layout — lanes 0-15: N=lr, M=r; lanes 16-31: M=8+r
    const int rb = (lane < 16) ? 0 : 8;
#pragma unroll
    for (int sm = 0; sm < 2; ++sm)
#pragma unroll
        for (int sn = 0; sn < 4; ++sn) {
            int gm = bm + waveM * 32 + sm * 16 + rb;
            int gn = bn + waveN * 64 + sn * 16 + lr;
#pragma unroll
            for (int r = 0; r < 8; ++r)
                C[(size_t)(gm + r) * ldc + gn] = acc[sm][sn][r];
        }
}

// ---------------------------------------------------------------------------
// Elementwise / reduction helpers
// ---------------------------------------------------------------------------
__global__ void init_scalars_kernel(float* s) {
    if (threadIdx.x < 16) s[threadIdx.x] = 0.f;
}

__global__ __launch_bounds__(256)
void rownormsq_kernel(const float* __restrict__ A, int cols, float* __restrict__ out) {
    __shared__ float red[256];
    const int row = blockIdx.x, tid = threadIdx.x;
    float s = 0.f;
    for (int c = tid; c < cols; c += 256) {
        float v = A[(size_t)row * cols + c];
        s += v * v;
    }
    red[tid] = s; __syncthreads();
    for (int st = 128; st > 0; st >>= 1) {
        if (tid < st) red[tid] += red[tid + st];
        __syncthreads();
    }
    if (tid == 0) out[row] = red[0];
}

__global__ __launch_bounds__(256)
void cvt_bf16_kernel(const float* __restrict__ src, bf16_t* __restrict__ dst) {
    size_t i = (size_t)blockIdx.x * 256 + threadIdx.x;
    dst[i] = f2bf(src[i]);
}

// Tiled f32 -> bf16 transpose of a BDIMxBDIM matrix (for T^T each iteration)
__global__ __launch_bounds__(256)
void transpose_cvt_kernel(const float* __restrict__ src, bf16_t* __restrict__ dstT) {
    __shared__ float tile[32][33];
    const int bx = blockIdx.x * 32, by = blockIdx.y * 32;
    const int tx = threadIdx.x & 31, ty = threadIdx.x >> 5;   // 32 x 8
#pragma unroll
    for (int r = 0; r < 32; r += 8)
        tile[ty + r][tx] = src[(size_t)(by + ty + r) * BDIM + bx + tx];
    __syncthreads();
#pragma unroll
    for (int r = 0; r < 32; r += 8)
        dstT[(size_t)(bx + ty + r) * BDIM + by + tx] = f2bf(tile[tx][ty + r]);
}

__global__ __launch_bounds__(256)
void copy_f32_kernel(const float* __restrict__ src, float* __restrict__ dst) {
    size_t i = (size_t)blockIdx.x * 256 + threadIdx.x;
    dst[i] = src[i];
}

__global__ __launch_bounds__(256)
void finish_pdist_kernel(const float* __restrict__ G, const float* __restrict__ sq,
                         float* __restrict__ Cout, float* __restrict__ sumslot) {
    __shared__ float red[256];
    const int tid = threadIdx.x;
    size_t i = (size_t)blockIdx.x * 256 + tid;
    int r = (int)(i >> 11), c = (int)(i & (BDIM - 1));
    float v = fmaxf(sq[r] + sq[c] - 2.f * G[i], 0.f);
    Cout[i] = v;
    red[tid] = v; __syncthreads();
    for (int st = 128; st > 0; st >>= 1) {
        if (tid < st) red[tid] += red[tid + st];
        __syncthreads();
    }
    if (tid == 0) atomicAdd(sumslot, red[0]);
}

__global__ __launch_bounds__(256)
void normalize_cvt_kernel(float* __restrict__ Cm, bf16_t* __restrict__ Cb,
                          const float* __restrict__ sumslot) {
    size_t i = (size_t)blockIdx.x * 256 + threadIdx.x;
    float scale = 1.f / (sumslot[0] * (1.f / ((float)BDIM * BDIM)) + 1e-8f);
    float v = Cm[i] * scale;
    Cm[i] = v;
    Cb[i] = f2bf(v);
}

__global__ __launch_bounds__(256)
void const_vec_kernel(const float* __restrict__ Cm, float* __restrict__ cvec,
                      float* __restrict__ sumsq_slot) {
    __shared__ float red[256];
    const int row = blockIdx.x, tid = threadIdx.x;
    float s = 0.f;
#pragma unroll
    for (int t = 0; t < 8; ++t) {
        float v = Cm[(size_t)row * BDIM + tid + t * 256];
        s += v * v;
    }
    red[tid] = s; __syncthreads();
    for (int st = 128; st > 0; st >>= 1) {
        if (tid < st) red[tid] += red[tid + st];
        __syncthreads();
    }
    if (tid == 0) {
        cvec[row] = red[0] * (1.f / (float)BDIM);
        atomicAdd(sumsq_slot, red[0]);
    }
}

__global__ __launch_bounds__(256)
void fill_bf16_kernel(bf16_t* __restrict__ dst, float val) {
    size_t i = (size_t)blockIdx.x * 256 + threadIdx.x;
    dst[i] = f2bf(val);
}

// expo = -(cx[i] + cz[j] - 2*cross); K = exp(expo - rowmax). In place. EPSILON=1.
__global__ __launch_bounds__(256)
void expo_rowmax_kernel(float* __restrict__ Km, const float* __restrict__ cx,
                        const float* __restrict__ cz) {
    __shared__ float red[256];
    const int row = blockIdx.x, tid = threadIdx.x;
    const float cxi = cx[row];
    float v[8]; float m = -3.4e38f;
#pragma unroll
    for (int t = 0; t < 8; ++t) {
        int j = tid + t * 256;
        float e = -(cxi + cz[j] - 2.f * Km[(size_t)row * BDIM + j]);
        v[t] = e; m = fmaxf(m, e);
    }
    red[tid] = m; __syncthreads();
    for (int st = 128; st > 0; st >>= 1) {
        if (tid < st) red[tid] = fmaxf(red[tid], red[tid + st]);
        __syncthreads();
    }
    m = red[0];
#pragma unroll
    for (int t = 0; t < 8; ++t)
        Km[(size_t)row * BDIM + tid + t * 256] = __expf(v[t] - m);
}

__global__ __launch_bounds__(256)
void row_normalize_kernel(float* __restrict__ Km) {
    __shared__ float red[256];
    const int row = blockIdx.x, tid = threadIdx.x;
    float v[8]; float s = 0.f;
#pragma unroll
    for (int t = 0; t < 8; ++t) {
        v[t] = Km[(size_t)row * BDIM + tid + t * 256];
        s += v[t];
    }
    red[tid] = s; __syncthreads();
    for (int st = 128; st > 0; st >>= 1) {
        if (tid < st) red[tid] += red[tid + st];
        __syncthreads();
    }
    float sc = (1.f / (float)BDIM) / (red[0] + 1e-8f);
#pragma unroll
    for (int t = 0; t < 8; ++t)
        Km[(size_t)row * BDIM + tid + t * 256] = v[t] * sc;
}

__global__ __launch_bounds__(256)
void colsum_kernel(const float* __restrict__ Km, float* __restrict__ col) {
    int j = blockIdx.x * 256 + threadIdx.x;
    float s = 0.f;
    for (int i = 0; i < BDIM; ++i) s += Km[(size_t)i * BDIM + j];
    col[j] = s;
}

__global__ __launch_bounds__(256)
void colscale_kernel(float* __restrict__ Km, const float* __restrict__ col) {
    size_t i = (size_t)blockIdx.x * 256 + threadIdx.x;
    int j = (int)(i & (BDIM - 1));
    Km[i] = Km[i] / (col[j] + 1e-8f) * (1.f / (float)BDIM);
}

__global__ __launch_bounds__(256)
void dot_reduce_kernel(const float* __restrict__ A, const float* __restrict__ Bv,
                       float* __restrict__ slot) {
    __shared__ float red[256];
    const int tid = threadIdx.x;
    size_t i = (size_t)blockIdx.x * 256 + tid;
    red[tid] = A[i] * Bv[i]; __syncthreads();
    for (int st = 128; st > 0; st >>= 1) {
        if (tid < st) red[tid] += red[tid + st];
        __syncthreads();
    }
    if (tid == 0) atomicAdd(slot, red[0]);
}

__global__ __launch_bounds__(256)
void reg_kernel(const float* __restrict__ Czm, const float* __restrict__ y,
                float* __restrict__ slot) {
    __shared__ float red[256];
    const int tid = threadIdx.x;
    size_t i = (size_t)blockIdx.x * 256 + tid;
    int r = (int)(i >> 11), c = (int)(i & (BDIM - 1));
    float v = 0.f;
    if (r != c) {
        float yd = fabsf(y[r] - y[c]);
        float zd = fmaxf(Czm[i], 1e-4f);   // sqrt(clip(Cz^2,1e-8)) with Cz>=0
        float d = __logf(yd + 1e-6f) - __logf(zd + 1e-6f);
        v = d * d;
    }
    red[tid] = v; __syncthreads();
    for (int st = 128; st > 0; st >>= 1) {
        if (tid < st) red[tid] += red[tid + st];
        __syncthreads();
    }
    if (tid == 0) atomicAdd(slot, red[0]);
}

__global__ void final_kernel(const float* __restrict__ s, float* __restrict__ out) {
    const float invb2 = 1.f / ((float)BDIM * BDIM);
    float gw = s[2] * invb2 + s[3] * invb2 - 2.f * s[4];
    gw = fmaxf(gw, 0.f);
    out[0] = gw + s[5] * (1.f / ((float)BDIM * (float)(BDIM - 1)));
}

// ---------------------------------------------------------------------------
extern "C" void kernel_launch(void* const* d_in, const int* in_sizes, int n_in,
                              void* d_out, int out_size, void* d_ws, size_t ws_size,
                              hipStream_t stream) {
    (void)in_sizes; (void)n_in; (void)out_size; (void)ws_size;
    const float* x = (const float*)d_in[0];   // 2048 x 256
    const float* z = (const float*)d_in[1];   // 2048 x 64
    const float* y = (const float*)d_in[2];   // 2048
    float* out = (float*)d_out;

    const size_t NB = (size_t)BDIM * BDIM;
    char* p = (char*)d_ws;
    float*  Cx  = (float*)p;  p += NB * 4;
    float*  Cz  = (float*)p;  p += NB * 4;
    float*  Tf  = (float*)p;  p += NB * 4;
    float*  M1  = (float*)p;  p += NB * 4;
    float*  Km  = (float*)p;  p += NB * 4;
    bf16_t* Cxb = (bf16_t*)p; p += NB * 2;
    bf16_t* Czb = (bf16_t*)p; p += NB * 2;
    bf16_t* Tt  = (bf16_t*)p; p += NB * 2;   // T^T in bf16 (B operand of gemm1)
    bf16_t* M1b = (bf16_t*)p; p += NB * 2;
    bf16_t* xb  = (bf16_t*)p; p += (size_t)BDIM * 256 * 2;
    bf16_t* zb  = (bf16_t*)p; p += (size_t)BDIM * 64 * 2;
    float*  sqx = (float*)p;  p += BDIM * 4;
    float*  sqz = (float*)p;  p += BDIM * 4;
    float*  cstx= (float*)p;  p += BDIM * 4;
    float*  cstz= (float*)p;  p += BDIM * 4;
    float*  colr= (float*)p;  p += BDIM * 4;
    float*  scal= (float*)p;  p += 64;

    dim3 blk(256);
    dim3 gemmGrid(BDIM / 128, BDIM / 128);
    dim3 trGrid(BDIM / 32, BDIM / 32);
    const int rowGrid = BDIM;
    const int ewGrid  = (int)(NB / 256);

    init_scalars_kernel<<<1, 32, 0, stream>>>(scal);

    // --- cost matrices (Gram via C = A * A^T -> Bt operand is A itself) ---
    rownormsq_kernel<<<rowGrid, blk, 0, stream>>>(x, 256, sqx);
    rownormsq_kernel<<<rowGrid, blk, 0, stream>>>(z, 64, sqz);
    cvt_bf16_kernel<<<(BDIM * 256) / 256, blk, 0, stream>>>(x, xb);
    cvt_bf16_kernel<<<(BDIM * 64) / 256, blk, 0, stream>>>(z, zb);

    wmma_gemm_bf16<<<gemmGrid, blk, 0, stream>>>(xb, xb, M1, 256, 256, 256, BDIM);
    finish_pdist_kernel<<<ewGrid, blk, 0, stream>>>(M1, sqx, Cx, scal + 0);
    normalize_cvt_kernel<<<ewGrid, blk, 0, stream>>>(Cx, Cxb, scal + 0);
    const_vec_kernel<<<rowGrid, blk, 0, stream>>>(Cx, cstx, scal + 2);

    wmma_gemm_bf16<<<gemmGrid, blk, 0, stream>>>(zb, zb, M1, 64, 64, 64, BDIM);
    finish_pdist_kernel<<<ewGrid, blk, 0, stream>>>(M1, sqz, Cz, scal + 1);
    normalize_cvt_kernel<<<ewGrid, blk, 0, stream>>>(Cz, Czb, scal + 1);
    const_vec_kernel<<<rowGrid, blk, 0, stream>>>(Cz, cstz, scal + 3);

    // --- T0 = 1/B (constant matrix is its own transpose) ---
    fill_bf16_kernel<<<ewGrid, blk, 0, stream>>>(Tt, 1.f / (float)BDIM);

    // --- Sinkhorn loop; C_x, C_z symmetric so C_z^T == C_z ---
    for (int it = 0; it < NITER; ++it) {
        wmma_gemm_bf16<<<gemmGrid, blk, 0, stream>>>(Cxb, Tt, M1, BDIM, BDIM, BDIM, BDIM);
        cvt_bf16_kernel<<<ewGrid, blk, 0, stream>>>(M1, M1b);
        wmma_gemm_bf16<<<gemmGrid, blk, 0, stream>>>(M1b, Czb, Km, BDIM, BDIM, BDIM, BDIM);
        expo_rowmax_kernel<<<rowGrid, blk, 0, stream>>>(Km, cstx, cstz);
        for (int n = 0; n < NNORM; ++n) {
            row_normalize_kernel<<<rowGrid, blk, 0, stream>>>(Km);
            colsum_kernel<<<BDIM / 256, blk, 0, stream>>>(Km, colr);
            colscale_kernel<<<ewGrid, blk, 0, stream>>>(Km, colr);
        }
        transpose_cvt_kernel<<<trGrid, blk, 0, stream>>>(Km, Tt);  // Tt = (T_new)^T
    }

    // --- final cross term: sum(C_x @ T @ C_z^T * T) ---
    copy_f32_kernel<<<ewGrid, blk, 0, stream>>>(Km, Tf);
    wmma_gemm_bf16<<<gemmGrid, blk, 0, stream>>>(Cxb, Tt, M1, BDIM, BDIM, BDIM, BDIM);
    cvt_bf16_kernel<<<ewGrid, blk, 0, stream>>>(M1, M1b);
    wmma_gemm_bf16<<<gemmGrid, blk, 0, stream>>>(M1b, Czb, Km, BDIM, BDIM, BDIM, BDIM);
    dot_reduce_kernel<<<ewGrid, blk, 0, stream>>>(Km, Tf, scal + 4);

    // --- regularizer + final assembly ---
    reg_kernel<<<ewGrid, blk, 0, stream>>>(Cz, y, scal + 5);
    final_kernel<<<1, 1, 0, stream>>>(scal, out);
}
